// MultiHeadAttention_446676599173
// MI455X (gfx1250) — matrix operand order repaired and checked by reading, not compiled, not run on hardware
//
#include <hip/hip_runtime.h>

// ---------------------------------------------------------------------------
// Problem constants (from reference)
// ---------------------------------------------------------------------------
constexpr int EMB   = 768;
constexpr int QKV3  = 3 * EMB;       // 2304
constexpr int HEADS = 12;
constexpr int DH    = 64;            // head dim
constexpr int BATCH = 8;
constexpr int NSEQ  = 1024;
constexpr int BN    = BATCH * NSEQ;  // 8192 rows

// ---------------------------------------------------------------------------
// WMMA helper types / fragment loaders (CDNA5 wave32, 16x16x32 f16 -> f32)
//   A (16x32 f16): lane m=lane&15, half=lane>>4; elems 0..7 -> K=half*8+j,
//                  elems 8..15 -> K=16+half*8+(j-8)   (two aligned b128 loads)
//   B (32x16 f16): lane n=lane&15, half=lane>>4; same K order as A
//                  (loaded from an [n][k] transposed tile: two b128 loads)
//   C/D (16x16 f32): VGPR r, lane l -> row = r + 8*(l>>4), col = l&15
// ---------------------------------------------------------------------------
typedef __attribute__((ext_vector_type(16))) _Float16 v16h;
typedef __attribute__((ext_vector_type(8)))  _Float16 v8h;
typedef __attribute__((ext_vector_type(8)))  float    v8f;

__device__ __forceinline__ v8f zero8() {
  v8f z = {0.f, 0.f, 0.f, 0.f, 0.f, 0.f, 0.f, 0.f};
  return z;
}

__device__ __forceinline__ v16h load_frag_a(const _Float16* __restrict__ base, int ld) {
  const int lane = threadIdx.x & 31;
  const int half = lane >> 4;
  const int m    = lane & 15;
  const _Float16* p = base + m * ld + half * 8;
  v8h lo = *(const v8h*)(p);
  v8h hi = *(const v8h*)(p + 16);
  return __builtin_shufflevector(lo, hi, 0, 1, 2, 3, 4, 5, 6, 7,
                                         8, 9, 10, 11, 12, 13, 14, 15);
}

// baseT indexed [n][k] (B stored transposed, row stride ld)
__device__ __forceinline__ v16h load_frag_bt(const _Float16* __restrict__ baseT, int ld) {
  const int lane = threadIdx.x & 31;
  const int half = lane >> 4;
  const int n    = lane & 15;
  const _Float16* p = baseT + n * ld + half * 8;
  v8h lo = *(const v8h*)(p);
  v8h hi = *(const v8h*)(p + 16);
  return __builtin_shufflevector(lo, hi, 0, 1, 2, 3, 4, 5, 6, 7,
                                         8, 9, 10, 11, 12, 13, 14, 15);
}

__device__ __forceinline__ v8f wmma16(v16h a, v16h b, v8f c) {
  return __builtin_amdgcn_wmma_f32_16x16x32_f16(false, a, false, b,
                                                (short)0, c, false, false);
}

// ---------------------------------------------------------------------------
// Async global->LDS staging (CDNA5 GLOBAL_LOAD_ASYNC_TO_LDS_B128, ASYNCcnt).
// Builtin param 0 is 'int __vector_size__(16) __device__ *' (addrspace(1),
// non-const, gcc-style vector) per hipcc diagnostics; param 1 is the LDS
// destination in addrspace(3). Guarded: falls back to a synchronous copy.
// ---------------------------------------------------------------------------
#if __has_builtin(__builtin_amdgcn_global_load_async_to_lds_b128)
#define HAVE_ASYNC_LDS 1
#else
#define HAVE_ASYNC_LDS 0
#endif

typedef int v4i_gcc __attribute__((vector_size(16)));
typedef __attribute__((address_space(1))) v4i_gcc* as1_v4i;
typedef __attribute__((address_space(3))) v4i_gcc* as3_v4i;

__device__ __forceinline__ void async_copy_b128(const _Float16* __restrict__ g,
                                                _Float16* __restrict__ l) {
#if HAVE_ASYNC_LDS
  __builtin_amdgcn_global_load_async_to_lds_b128((as1_v4i)g, (as3_v4i)l, 0, 0);
#else
  *(v8h*)l = *(const v8h*)g;
#endif
}

#if HAVE_ASYNC_LDS
#if __has_builtin(__builtin_amdgcn_s_wait_asynccnt)
#define WAIT_ASYNC(n) __builtin_amdgcn_s_wait_asynccnt(n)
#else
#define WAIT_ASYNC(n) asm volatile("s_wait_asynccnt %0" ::"n"(n) : "memory")
#endif
#else
#define WAIT_ASYNC(n) do {} while (0)
#endif

// stage a 128x32 f16 tile (contiguous rows, row stride ldg) with 256 threads:
// 512 b128 chunks, 2 per thread.
__device__ __forceinline__ void stage_128x32(const _Float16* __restrict__ g,
                                             int row0, int ldg, int k0,
                                             _Float16* __restrict__ l, int tid) {
#pragma unroll
  for (int u = 0; u < 2; ++u) {
    const int ci = tid + u * 256;
    const int r  = ci >> 2;
    const int c  = (ci & 3) * 8;
    async_copy_b128(g + (size_t)(row0 + r) * ldg + k0 + c, l + r * 32 + c);
  }
}

// stage a 64x64 f16 tile (g pre-offset to tile origin) with 128 threads:
// 512 b128 chunks, 4 per thread.
__device__ __forceinline__ void stage_64x64(const _Float16* __restrict__ g,
                                            size_t rowstride,
                                            _Float16* __restrict__ l, int tid) {
#pragma unroll
  for (int u = 0; u < 4; ++u) {
    const int ci = tid + u * 128;
    const int r  = ci >> 3;
    const int c  = (ci & 7) * 8;
    async_copy_b128(g + (size_t)r * rowstride + c, l + r * 64 + c);
  }
}

// ---------------------------------------------------------------------------
// Kernel 1a: fp32 -> f16 conversion (grid-stride)
// ---------------------------------------------------------------------------
__global__ void cvt_f32_to_f16_kernel(const float* __restrict__ src,
                                      _Float16* __restrict__ dst, int n) {
  for (int i = blockIdx.x * blockDim.x + threadIdx.x; i < n;
       i += gridDim.x * blockDim.x)
    dst[i] = (_Float16)src[i];
}

// Kernel 1b: fp32 [R][C] -> f16 transposed [C][R]  (one-time; hoists all
// B-operand transposes out of the GEMM hot loops)
__global__ void cvt_transpose_f16_kernel(const float* __restrict__ src,
                                         _Float16* __restrict__ dst, int R,
                                         int C) {
  const int total = R * C;
  for (int i = blockIdx.x * blockDim.x + threadIdx.x; i < total;
       i += gridDim.x * blockDim.x) {
    const int r = i / C;
    const int c = i - r * C;
    dst[(size_t)c * R + r] = (_Float16)src[i];
  }
}

// ---------------------------------------------------------------------------
// Kernel 2: QKV GEMM  C(8192x2304) = X(8192x768) @ W + b, W given transposed
// as WT[2304][768]. 256 threads = 8 waves, tile 128x128, K-step 32, async
// double-buffered staging. Epilogue scatters Q,K as (b,h,n,d) and V
// TRANSPOSED as (b,h,d,n) so the attention V tile is a contiguous copy.
// ---------------------------------------------------------------------------
__global__ void __launch_bounds__(256) qkv_gemm_kernel(
    const _Float16* __restrict__ X, const _Float16* __restrict__ WT,
    const float* __restrict__ bias, _Float16* __restrict__ Qh,
    _Float16* __restrict__ Kh, _Float16* __restrict__ Vt) {
  __shared__ __align__(16) _Float16 Xs[2][128 * 32];  // [m][k]
  __shared__ __align__(16) _Float16 Ws[2][128 * 32];  // [n][k]

  const int tid  = threadIdx.x;
  const int wave = tid >> 5;
  const int lane = tid & 31;
  const int half = lane >> 4;
  const int nlo  = lane & 15;
  const int m0 = blockIdx.y * 128;
  const int n0 = blockIdx.x * 128;
  const int wm = (wave & 1) * 64;   // 2 waves along M
  const int wn = (wave >> 1) * 32;  // 4 waves along N

  v8f acc[4][2];
#pragma unroll
  for (int i = 0; i < 4; ++i)
#pragma unroll
    for (int j = 0; j < 2; ++j) acc[i][j] = zero8();

  constexpr int KT = EMB / 32;  // 24 K-steps
  stage_128x32(X, m0, EMB, 0, Xs[0], tid);
  stage_128x32(WT, n0, EMB, 0, Ws[0], tid);

  for (int kt = 0; kt < KT; ++kt) {
    const int cur = kt & 1;
    if (kt + 1 < KT) {
      stage_128x32(X, m0, EMB, (kt + 1) * 32, Xs[cur ^ 1], tid);
      stage_128x32(WT, n0, EMB, (kt + 1) * 32, Ws[cur ^ 1], tid);
      WAIT_ASYNC(4);  // 4 newest outstanding = next tile; current complete
    } else {
      WAIT_ASYNC(0);
    }
    __syncthreads();

    v16h af[4];
    v16h bf[2];
#pragma unroll
    for (int i = 0; i < 4; ++i)
      af[i] = load_frag_a(Xs[cur] + (wm + 16 * i) * 32, 32);
#pragma unroll
    for (int j = 0; j < 2; ++j)
      bf[j] = load_frag_bt(Ws[cur] + (wn + 16 * j) * 32, 32);
#pragma unroll
    for (int i = 0; i < 4; ++i)
#pragma unroll
      for (int j = 0; j < 2; ++j) acc[i][j] = wmma16(af[i], bf[j], acc[i][j]);
    __syncthreads();  // all reads done before this buffer is refilled
  }

  // epilogue: add bias, scatter to Q/K (b,h,n,d) and V (b,h,d,n) as f16
#pragma unroll
  for (int i = 0; i < 4; ++i)
#pragma unroll
    for (int j = 0; j < 2; ++j)
#pragma unroll
      for (int r = 0; r < 8; ++r) {
        const int row = m0 + wm + 16 * i + r + 8 * half;
        const int col = n0 + wn + 16 * j + nlo;
        const float v = acc[i][j][r] + bias[col];
        const int s   = col % 3;  // qkv is the innermost factor
        const int rem = col / 3;
        const int dd  = rem & 63;
        const int hh  = rem >> 6;
        const int bb  = row >> 10;
        const int nn  = row & 1023;
        const size_t bhb = (size_t)(bb * HEADS + hh);
        const _Float16 hv = (_Float16)v;
        if (s == 0)      Qh[(bhb * NSEQ + nn) * DH + dd] = hv;
        else if (s == 1) Kh[(bhb * NSEQ + nn) * DH + dd] = hv;
        else             Vt[(bhb * DH + dd) * NSEQ + nn] = hv;
      }
}

// ---------------------------------------------------------------------------
// Kernel 3: flash attention per (b*h, 64-row q tile); 128 threads = 4 waves.
// energy unscaled; online softmax; final scale 1/(l * sqrt(768)).
// All tiles staged with async copies; V arrives pre-transposed.
// ---------------------------------------------------------------------------
__global__ void __launch_bounds__(128) attn_kernel(
    const _Float16* __restrict__ Qh, const _Float16* __restrict__ Kh,
    const _Float16* __restrict__ Vt, _Float16* __restrict__ Ah) {
  __shared__ __align__(16) _Float16 Qs[64 * 64];   // [q][d]
  __shared__ __align__(16) _Float16 Ks[64 * 64];   // [key][d]
  __shared__ __align__(16) _Float16 VsT[64 * 64];  // [d][key]
  __shared__ __align__(16) float    Sls[64 * 64];  // scores f32
  __shared__ __align__(16) _Float16 Pls[64 * 64];  // exp probs f16 [q][key]
  __shared__ float rowm[64], rowl[64], rowf[64];

  const int tid  = threadIdx.x;
  const int wave = tid >> 5;
  const int lane = tid & 31;
  const int half = lane >> 4;
  const int nlo  = lane & 15;
  const int bh = blockIdx.y;  // 0..95
  const int q0 = blockIdx.x * 64;
  const size_t base_bh = (size_t)bh * NSEQ * DH;
  const int rbase = wave * 16;  // 16 q-rows per wave

  stage_64x64(Qh + base_bh + (size_t)q0 * DH, DH, Qs, tid);
  if (tid < 64) {
    rowm[tid] = -1e30f;
    rowl[tid] = 0.f;
  }

  v8f o[4];
#pragma unroll
  for (int t = 0; t < 4; ++t) o[t] = zero8();

  for (int kt = 0; kt < NSEQ / 64; ++kt) {
    const int key0 = kt * 64;
    stage_64x64(Kh + base_bh + (size_t)key0 * DH, DH, Ks, tid);
    // Vt rows are d (stride NSEQ); take key columns [key0, key0+64)
    stage_64x64(Vt + base_bh + key0, NSEQ, VsT, tid);
    WAIT_ASYNC(0);
    __syncthreads();

    // S = Q Kt : this wave's 16 rows x 64 keys
#pragma unroll
    for (int t = 0; t < 4; ++t) {
      v8f s = zero8();
#pragma unroll
      for (int ks = 0; ks < DH; ks += 32) {
        v16h a = load_frag_a(Qs + rbase * 64 + ks, 64);
        v16h b = load_frag_bt(Ks + (t * 16) * 64 + ks, 64);
        s = wmma16(a, b, s);
      }
#pragma unroll
      for (int r = 0; r < 8; ++r)
        Sls[(rbase + r + 8 * half) * 64 + t * 16 + nlo] = s[r];
    }
    __syncthreads();

    // online softmax (one thread per q row)
    if (tid < 64) {
      const int row = tid;
      const float mold = rowm[row];
      float mx = mold;
      for (int c = 0; c < 64; ++c) mx = fmaxf(mx, Sls[row * 64 + c]);
      const float f = __expf(mold - mx);
      float sum = 0.f;
      for (int c = 0; c < 64; ++c) {
        const float p = __expf(Sls[row * 64 + c] - mx);
        Pls[row * 64 + c] = (_Float16)p;
        sum += p;
      }
      rowl[row] = rowl[row] * f + sum;
      rowm[row] = mx;
      rowf[row] = f;
    }
    __syncthreads();

    // rescale O, then O += P @ V
#pragma unroll
    for (int t = 0; t < 4; ++t)
#pragma unroll
      for (int r = 0; r < 8; ++r) o[t][r] *= rowf[rbase + r + 8 * half];
#pragma unroll
    for (int ks = 0; ks < 64; ks += 32) {
      v16h a = load_frag_a(Pls + rbase * 64 + ks, 64);
#pragma unroll
      for (int t = 0; t < 4; ++t) {
        v16h b = load_frag_bt(VsT + (t * 16) * 64 + ks, 64);
        o[t] = wmma16(a, b, o[t]);
      }
    }
    __syncthreads();  // before Ks/VsT/Pls are overwritten
  }

  // epilogue: out = O / (l * sqrt(EMB)); write (b, n, h*64+d) as f16
  const int bb = bh / HEADS;
  const int hh = bh % HEADS;
#pragma unroll
  for (int t = 0; t < 4; ++t)
#pragma unroll
    for (int r = 0; r < 8; ++r) {
      const int row = rbase + r + 8 * half;
      const float inv = 1.0f / (rowl[row] * 27.712812921102035f);
      const int n = q0 + row;
      const int d = t * 16 + nlo;
      Ah[((size_t)(bb * NSEQ + n)) * EMB + hh * DH + d] =
          (_Float16)(o[t][r] * inv);
    }
}

// ---------------------------------------------------------------------------
// Kernel 4: projection GEMM  Out(8192x768) = A(8192x768) @ Wp + b,
// Wp given transposed as WT[768][768]. fp32 output straight into d_out.
// ---------------------------------------------------------------------------
__global__ void __launch_bounds__(256) proj_gemm_kernel(
    const _Float16* __restrict__ A, const _Float16* __restrict__ WT,
    const float* __restrict__ bias, float* __restrict__ Out) {
  __shared__ __align__(16) _Float16 As[2][128 * 32];
  __shared__ __align__(16) _Float16 Ws[2][128 * 32];

  const int tid  = threadIdx.x;
  const int wave = tid >> 5;
  const int lane = tid & 31;
  const int half = lane >> 4;
  const int nlo  = lane & 15;
  const int m0 = blockIdx.y * 128;
  const int n0 = blockIdx.x * 128;
  const int wm = (wave & 1) * 64;
  const int wn = (wave >> 1) * 32;

  v8f acc[4][2];
#pragma unroll
  for (int i = 0; i < 4; ++i)
#pragma unroll
    for (int j = 0; j < 2; ++j) acc[i][j] = zero8();

  constexpr int KT = EMB / 32;
  stage_128x32(A, m0, EMB, 0, As[0], tid);
  stage_128x32(WT, n0, EMB, 0, Ws[0], tid);

  for (int kt = 0; kt < KT; ++kt) {
    const int cur = kt & 1;
    if (kt + 1 < KT) {
      stage_128x32(A, m0, EMB, (kt + 1) * 32, As[cur ^ 1], tid);
      stage_128x32(WT, n0, EMB, (kt + 1) * 32, Ws[cur ^ 1], tid);
      WAIT_ASYNC(4);
    } else {
      WAIT_ASYNC(0);
    }
    __syncthreads();

    v16h af[4];
    v16h bf[2];
#pragma unroll
    for (int i = 0; i < 4; ++i)
      af[i] = load_frag_a(As[cur] + (wm + 16 * i) * 32, 32);
#pragma unroll
    for (int j = 0; j < 2; ++j)
      bf[j] = load_frag_bt(Ws[cur] + (wn + 16 * j) * 32, 32);
#pragma unroll
    for (int i = 0; i < 4; ++i)
#pragma unroll
      for (int j = 0; j < 2; ++j) acc[i][j] = wmma16(af[i], bf[j], acc[i][j]);
    __syncthreads();
  }

#pragma unroll
  for (int i = 0; i < 4; ++i)
#pragma unroll
    for (int j = 0; j < 2; ++j)
#pragma unroll
      for (int r = 0; r < 8; ++r) {
        const int row = m0 + wm + 16 * i + r + 8 * half;
        const int col = n0 + wn + 16 * j + nlo;
        Out[(size_t)row * EMB + col] = acc[i][j][r] + bias[col];
      }
}

// ---------------------------------------------------------------------------
// Host-side launcher
// ---------------------------------------------------------------------------
extern "C" void kernel_launch(void* const* d_in, const int* in_sizes, int n_in,
                              void* d_out, int out_size, void* d_ws,
                              size_t ws_size, hipStream_t stream) {
  (void)in_sizes; (void)n_in; (void)out_size; (void)ws_size;
  const float* x     = (const float*)d_in[0];
  const float* Wqkv  = (const float*)d_in[1];
  const float* bqkv  = (const float*)d_in[2];
  const float* Wproj = (const float*)d_in[3];
  const float* bproj = (const float*)d_in[4];
  float* out = (float*)d_out;

  // workspace layout (f16)
  _Float16* p = (_Float16*)d_ws;
  _Float16* Xh      = p; p += (size_t)BN * EMB;    // x in f16
  _Float16* WqkvT   = p; p += (size_t)EMB * QKV3;  // [2304][768]
  _Float16* WprojT  = p; p += (size_t)EMB * EMB;   // [768][768]
  _Float16* Qh      = p; p += (size_t)BN * EMB;    // (b,h,n,d)
  _Float16* Kh      = p; p += (size_t)BN * EMB;    // (b,h,n,d)
  _Float16* Vt      = p; p += (size_t)BN * EMB;    // (b,h,d,n)  transposed
  _Float16* Ah      = p; p += (size_t)BN * EMB;    // attn out (b,n,e)

  cvt_f32_to_f16_kernel<<<1024, 256, 0, stream>>>(x, Xh, BN * EMB);
  cvt_transpose_f16_kernel<<<512, 256, 0, stream>>>(Wqkv, WqkvT, EMB, QKV3);
  cvt_transpose_f16_kernel<<<256, 256, 0, stream>>>(Wproj, WprojT, EMB, EMB);

  qkv_gemm_kernel<<<dim3(QKV3 / 128, BN / 128), 256, 0, stream>>>(
      Xh, WqkvT, bqkv, Qh, Kh, Vt);

  attn_kernel<<<dim3(NSEQ / 64, BATCH * HEADS), 128, 0, stream>>>(
      Qh, Kh, Vt, Ah);

  proj_gemm_kernel<<<dim3(EMB / 128, BN / 128), 256, 0, stream>>>(
      Ah, WprojT, bproj, out);
}
